// GCN_21371757265570
// MI455X (gfx1250) — matrix operand order repaired and checked by reading, not compiled, not run on hardware
//
#include <hip/hip_runtime.h>
#include <hip/hip_bf16.h>

typedef __bf16 v16bf __attribute__((ext_vector_type(16)));
typedef __bf16 v8bf  __attribute__((ext_vector_type(8)));
typedef float  v8f   __attribute__((ext_vector_type(8)));
typedef float  v4f   __attribute__((ext_vector_type(4)));
typedef unsigned int v4u __attribute__((ext_vector_type(4)));
typedef int v8i __attribute__((ext_vector_type(8)));
typedef int v4i __attribute__((ext_vector_type(4)));

#define NNODES 16384
#define HDIM   128
#define KC     64              // k-chunk staged in LDS per step
#define KSTEPS (KC / 32)       // wmma k-steps per chunk
#define LPITCH 72              // LDS row pitch in bf16 elems (144 B: padded, bank-friendly)
#define NCHUNK (NNODES / KC)
#define PFD    1024            // adj prefetch distance in floats (4 KB/row, 32 k-steps ahead)

#if __has_builtin(__builtin_amdgcn_tensor_load_to_lds)
#define HAVE_TDM 1
#else
#define HAVE_TDM 0
#endif

// ---- non-temporal adj stream loads (adj lines are consumed exactly once) ----
__device__ inline v4f ld_nt4(const float* p) {
    return __builtin_nontemporal_load((const v4f*)p);
}

// ---- A-operand: build v16bf in WMMA 16x32 A layout ----
__device__ inline v16bf cvt_a(v4f a0, v4f a1, v4f a2, v4f a3) {
    v16bf a;
#pragma unroll
    for (int e = 0; e < 4; ++e) {
        a[e]      = (__bf16)a0[e];
        a[e + 4]  = (__bf16)a1[e];
        a[e + 8]  = (__bf16)a2[e];
        a[e + 12] = (__bf16)a3[e];
    }
    return a;
}

__device__ inline v16bf load_a(const float* __restrict__ arow, int k0, int half) {
    const float* p0 = arow + k0 + half * 8;
    const float* p1 = arow + k0 + 16 + half * 8;
    return cvt_a(*(const v4f*)(p0), *(const v4f*)(p0 + 4),
                 *(const v4f*)(p1), *(const v4f*)(p1 + 4));
}

__device__ inline v16bf load_a(const __bf16* __restrict__ arow, int k0, int half) {
    v8bf lo = *(const v8bf*)(arow + k0 + half * 8);
    v8bf hi = *(const v8bf*)(arow + k0 + 16 + half * 8);
    v16bf a;
#pragma unroll
    for (int e = 0; e < 8; ++e) { a[e] = lo[e]; a[e + 8] = hi[e]; }
    return a;
}

__device__ inline v16bf lds_b16(const __bf16* p) {   // 32B from LDS, 16B-aligned safe
    v8bf b0 = *(const v8bf*)(p);
    v8bf b1 = *(const v8bf*)(p + 16);
    v16bf b;
#pragma unroll
    for (int e = 0; e < 8; ++e) { b[e] = b0[e]; b[e + 8] = b1[e]; }
    return b;
}

#if HAVE_TDM
// TDM: DMA a [128 rows x KC cols] bf16 tile of SbT into LDS with padded pitch.
__device__ inline void tdm_load_tile(const __bf16* tileStart, unsigned ldsByteOff) {
    unsigned long long ga = (unsigned long long)(size_t)tileStart;
    v4u g0;
    g0[0] = 1u;                                   // count=1 (valid), user mode
    g0[1] = ldsByteOff;                           // lds_addr (bytes)
    g0[2] = (unsigned)(ga & 0xFFFFFFFFu);         // global_addr[31:0]
    g0[3] = (unsigned)((ga >> 32) & 0x01FFFFFFu)  // global_addr[56:32]
          | (2u << 30);                           // type = 2 (image/tensor)
    v8i g1;
    g1[0] = (int)((1u << 16)      // data_size = 2 bytes
                | (1u << 20)      // pad_enable
                | (4u << 22)      // pad_interval: 32 DWORDs (=128B = one tile row)
                | (3u << 25));    // pad_amount: 4 DWORDs (=16B) -> 144B pitch
    const unsigned dim0 = NNODES, dim1 = 128, tile0 = KC, tile1 = 128;
    g1[1] = (int)((dim0 & 0xFFFFu) << 16);                       // tensor_dim0[15:0]
    g1[2] = (int)(((dim0 >> 16) & 0xFFFFu) | ((dim1 & 0xFFFFu) << 16));
    g1[3] = (int)(((dim1 >> 16) & 0xFFFFu) | (tile0 << 16));     // tile_dim0
    g1[4] = (int)(tile1);                                        // tile_dim1, tile_dim2=0
    g1[5] = (int)(NNODES);                                       // tensor_dim0_stride lo
    g1[6] = 0;
    g1[7] = 0;
    v4i g2 = {0, 0, 0, 0};
    v4i g3 = {0, 0, 0, 0};
    v8i g4 = {0, 0, 0, 0, 0, 0, 0, 0};
    __builtin_amdgcn_tensor_load_to_lds(g0, g1, g2, g3, g4, 0);
}
#endif

// ---- fp32 W[k][n] -> bf16 WT[n][k] ----
__global__ __launch_bounds__(256) void cvt_transpose_kernel(
    const float* __restrict__ W, __bf16* __restrict__ WT, int K, int Ncols) {
    int idx = blockIdx.x * 256 + threadIdx.x;
    if (idx < K * Ncols) {
        int k = idx / Ncols;
        int n = idx - k * Ncols;
        WT[(size_t)n * K + k] = (__bf16)W[idx];
    }
}

// ---- small GEMM: SbT[n][m] = bf16( A[m][k] @ BT[n][k]^T ), M=16384, N=128 ----
template <typename AT>
__global__ __launch_bounds__(256, 1) void gcn_small_gemm(
    const AT* __restrict__ A, int K,
    const __bf16* __restrict__ BT,   // [128][K]
    __bf16* __restrict__ SbT)        // [128][16384]
{
    const int wave = threadIdx.x >> 5;
    const int lane = threadIdx.x & 31;
    const int l15  = lane & 15;
    const int half = lane >> 4;
    const int row  = blockIdx.x * 128 + wave * 16 + l15;

    v8f acc[8];
#pragma unroll
    for (int j = 0; j < 8; ++j)
#pragma unroll
        for (int i = 0; i < 8; ++i) acc[j][i] = 0.0f;

    const AT* arow = A + (size_t)row * K;
    for (int k0 = 0; k0 < K; k0 += 32) {
        v16bf a = load_a(arow, k0, half);
#pragma unroll
        for (int j = 0; j < 8; ++j) {
            const __bf16* bp = BT + (size_t)(j * 16 + l15) * K + k0 + half * 8;
            v16bf b = lds_b16(bp);
            acc[j] = __builtin_amdgcn_wmma_f32_16x16x32_bf16(
                false, a, false, b, (short)0, acc[j], false, false);
        }
    }

    const int mBase = blockIdx.x * 128 + wave * 16 + half * 8;
#pragma unroll
    for (int j = 0; j < 8; ++j) {
        const int c = j * 16 + l15;
        v8bf o;
#pragma unroll
        for (int i = 0; i < 8; ++i) o[i] = (__bf16)acc[j][i];
        *(v8bf*)(SbT + (size_t)c * NNODES + mBase) = o;
    }
}

__device__ inline unsigned mix_hash(unsigned idx, unsigned seed) {
    unsigned h = idx ^ (seed * 0x9E3779B9u);
    h *= 0x85EBCA6Bu; h ^= h >> 13;
    h *= 0xC2B2AE35u; h ^= h >> 16;
    return h;
}

// ---- big GEMM: out = adj @ Sb + bias ; B tile staged in LDS (TDM double-buffer) ----
__global__ __launch_bounds__(256, 1) void gcn_adj_gemm(
    const float* __restrict__ adj,    // [16384][16384] fp32, streamed (NT + prefetch)
    const __bf16* __restrict__ SbT,   // [128][16384] bf16 (B transposed)
    const float* __restrict__ bias,   // [128]
    float* __restrict__ outF,         // [16384][128] fp32 (NT stores)
    __bf16* __restrict__ Xb,          // [16384][128] bf16 dropout out (or null)
    int relu, int dropout, unsigned seed)
{
    __shared__ __align__(16) __bf16 ldsB[2][128 * LPITCH];  // 2 x 18 KB

    const int wave = threadIdx.x >> 5;
    const int lane = threadIdx.x & 31;
    const int l15  = lane & 15;
    const int half = lane >> 4;
    const int row  = blockIdx.x * 128 + wave * 16 + l15;

    v8f acc[8];
#pragma unroll
    for (int j = 0; j < 8; ++j)
#pragma unroll
        for (int i = 0; i < 8; ++i) acc[j][i] = 0.0f;

    const float* arow = adj + (size_t)row * NNODES;

    // Warm the stream: prefetch the first PFD floats of this row (one line per step).
    for (int kp = 0; kp < PFD; kp += 32)
        __builtin_prefetch(arow + kp, 0, 0);

    // A prefetch registers (one k-step ahead), non-temporal stream
    v4f ar0 = ld_nt4(arow + half * 8);
    v4f ar1 = ld_nt4(arow + half * 8 + 4);
    v4f ar2 = ld_nt4(arow + 16 + half * 8);
    v4f ar3 = ld_nt4(arow + 16 + half * 8 + 4);

#if HAVE_TDM
    if (wave == 0) {
        tdm_load_tile(SbT, (unsigned)(size_t)&ldsB[0][0]);  // chunk 0 -> buf 0
    }
#endif

    for (int c = 0; c < NCHUNK; ++c) {
        const int buf = c & 1;
#if HAVE_TDM
        if (wave == 0) {
            if (c + 1 < NCHUNK) {
                tdm_load_tile(SbT + (c + 1) * KC, (unsigned)(size_t)&ldsB[buf ^ 1][0]);
                __builtin_amdgcn_s_wait_tensorcnt(1);
            } else {
                __builtin_amdgcn_s_wait_tensorcnt(0);
            }
        }
        __syncthreads();
#else
        // cooperative staging fallback: 256 threads copy 128 x KC bf16 tile
        {
            const int t = threadIdx.x;
            const int r = t >> 1, h = t & 1;
            const __bf16* src = SbT + (size_t)r * NNODES + c * KC + h * 32;
            __bf16* dst = &ldsB[buf][r * LPITCH + h * 32];
#pragma unroll
            for (int q = 0; q < 4; ++q)
                *(v8bf*)(dst + q * 8) = *(const v8bf*)(src + q * 8);
        }
        __syncthreads();
#endif

#pragma unroll
        for (int s = 0; s < KSTEPS; ++s) {
            v16bf a = cvt_a(ar0, ar1, ar2, ar3);
            const int kbase = c * KC + s * 32;
            // deep HBM prefetch: run the adj stream PFD floats ahead (no VGPR/LOADcnt cost)
            const int kpf = kbase + PFD;
            if (kpf < NNODES)                 // uniform; stays inside this adj row
                __builtin_prefetch(arow + kpf, 0, 0);
            const int knext = kbase + 32;
            if (knext < NNODES) {             // uniform: next k-step of adj into registers
                ar0 = ld_nt4(arow + knext + half * 8);
                ar1 = ld_nt4(arow + knext + half * 8 + 4);
                ar2 = ld_nt4(arow + knext + 16 + half * 8);
                ar3 = ld_nt4(arow + knext + 16 + half * 8 + 4);
            }
            const int sOff = s * 32 + half * 8;
#pragma unroll
            for (int j = 0; j < 8; ++j) {
                v16bf b = lds_b16(&ldsB[buf][(j * 16 + l15) * LPITCH + sOff]);
                acc[j] = __builtin_amdgcn_wmma_f32_16x16x32_bf16(
                    false, a, false, b, (short)0, acc[j], false, false);
            }
        }
        __syncthreads();
    }

    const int mBase = blockIdx.x * 128 + wave * 16 + half * 8;
#pragma unroll
    for (int j = 0; j < 8; ++j) {
        const int c = j * 16 + l15;
        const float bval = bias[c];
#pragma unroll
        for (int i = 0; i < 8; ++i) {
            const int m = mBase + i;
            float v = acc[j][i] + bval;
            if (relu) v = v > 0.0f ? v : 0.0f;
            __builtin_nontemporal_store(v, &outF[(size_t)m * HDIM + c]);
            if (dropout) {
                unsigned h = mix_hash((unsigned)(m * HDIM + c), seed);
                float dv = (h & 1u) ? v * 2.0f : 0.0f;
                Xb[(size_t)m * HDIM + c] = (__bf16)dv;   // cached: re-read next stage
            }
        }
    }
}

extern "C" void kernel_launch(void* const* d_in, const int* in_sizes, int n_in,
                              void* d_out, int out_size, void* d_ws, size_t ws_size,
                              hipStream_t stream) {
    const float* x   = (const float*)d_in[0];
    const float* adj = (const float*)d_in[1];
    const float* W1  = (const float*)d_in[2];
    const float* b1  = (const float*)d_in[3];
    const float* W2  = (const float*)d_in[4];
    const float* b2  = (const float*)d_in[5];
    const float* W3  = (const float*)d_in[6];
    const float* b3  = (const float*)d_in[7];
    float* out = (float*)d_out;

    char* ws = (char*)d_ws;
    __bf16* SbT = (__bf16*)(ws);                               // 128x16384 bf16 = 4 MB
    __bf16* Xb  = (__bf16*)(ws + (size_t)(4u << 20));          // 16384x128 bf16 = 4 MB
    __bf16* W1T = (__bf16*)(ws + (size_t)(8u << 20));          // 128x512  bf16 = 128 KB
    __bf16* W2T = (__bf16*)(ws + (size_t)(8u << 20) + (128u << 10));
    __bf16* W3T = (__bf16*)(ws + (size_t)(8u << 20) + (160u << 10));

    const size_t outStride = (size_t)NNODES * HDIM;

    cvt_transpose_kernel<<<(512 * 128) / 256, 256, 0, stream>>>(W1, W1T, 512, 128);
    cvt_transpose_kernel<<<(128 * 128) / 256, 256, 0, stream>>>(W2, W2T, 128, 128);
    cvt_transpose_kernel<<<(128 * 128) / 256, 256, 0, stream>>>(W3, W3T, 128, 128);

    gcn_small_gemm<float><<<NNODES / 128, 256, 0, stream>>>(x, 512, W1T, SbT);
    gcn_adj_gemm<<<NNODES / 128, 256, 0, stream>>>(adj, SbT, b1, out, Xb, 1, 1, 101u);
    gcn_small_gemm<__bf16><<<NNODES / 128, 256, 0, stream>>>(Xb, 128, W2T, SbT);
    gcn_adj_gemm<<<NNODES / 128, 256, 0, stream>>>(adj, SbT, b2, out + outStride, Xb, 0, 1, 202u);
    gcn_small_gemm<__bf16><<<NNODES / 128, 256, 0, stream>>>(Xb, 128, W3T, SbT);
    gcn_adj_gemm<<<NNODES / 128, 256, 0, stream>>>(adj, SbT, b3, out + 2 * outStride, nullptr, 0, 0, 0u);
}
